// BaselineSoftmax_14267881357581
// MI455X (gfx1250) — matrix operand order repaired and checked by reading, not compiled, not run on hardware
//
#include <hip/hip_runtime.h>

// Problem constants (match reference)
#define TOKENS 4096
#define DMODEL 2048
#define VOCAB  50257

// GEMM tiling
#define TM 256          // output tile rows per block   (weight re-read = TOKENS/TM = 16)
#define TN 64           // output tile cols per block
#define TK 32           // K chunk (bf16 WMMA K)
#define LDA 40          // padded LDS row stride (bf16 elems) -> 80B rows, conflict-free B128 reads
#define LDB 40

typedef __attribute__((ext_vector_type(16))) __bf16 v16bf;
typedef __attribute__((ext_vector_type(8)))  __bf16 v8bf;
typedef __attribute__((ext_vector_type(8)))  float  v8f;

__device__ __forceinline__ unsigned int pack_bf16x2(float a, float b) {
    unsigned int ua = __float_as_uint(a);
    unsigned int ub = __float_as_uint(b);
    ua = (ua + 0x7FFFu + ((ua >> 16) & 1u)) >> 16;   // round-to-nearest-even
    ub = (ub + 0x7FFFu + ((ub >> 16) & 1u)) >> 16;
    return (ua & 0xFFFFu) | (ub << 16);
}

// ---------------------------------------------------------------------------
// Kernel 1: logits = input @ W^T via v_wmma_f32_16x16x32_bf16, fp32 accumulate
// Block = 256 threads = 8 wave32, computes a 256(M) x 64(N) tile.
// Wave grid 4(M) x 2(N): each wave owns 64x32 = 4x2 WMMA 16x16 tiles.
// ---------------------------------------------------------------------------
__global__ __launch_bounds__(256)
void wmma_logits_kernel(const float* __restrict__ inp,   // [TOKENS, DMODEL]
                        const float* __restrict__ wgt,   // [VOCAB,  DMODEL]
                        float* __restrict__ out)         // [TOKENS, VOCAB]
{
    __shared__ __align__(16) unsigned short ash[TM * LDA];  // A tile, bf16 bits
    __shared__ __align__(16) unsigned short bsh[TN * LDB];  // B tile, bf16 bits ([vocab][k])

    const int tid  = threadIdx.x;
    const int lane = tid & 31;
    const int wv   = tid >> 5;        // 0..7
    const int m_w  = wv >> 1;         // 0..3 -> wave row base = m_w*64
    const int n_w  = wv & 1;          // 0..1 -> wave col base = n_w*32

    const int blockM = blockIdx.y * TM;
    const int blockN = blockIdx.x * TN;

    // fragment lane decode (ISA 7.12.2, 16-bit A 16x32 / C-D 16x16 layouts)
    const int hp  = lane >> 4;        // lane half: 0 or 1
    const int l16 = lane & 15;

    // global-load indexing (float4 granularity, 32 floats per tile row)
    const int c4 = tid & 7;           // float4 column within K-chunk
    const int r0 = tid >> 3;          // 0..31

    v8f acc[4][2] = {};

    for (int k0 = 0; k0 < DMODEL; k0 += TK) {
        // ---- A tile: 256 rows x 32 K, fp32 -> bf16 -> LDS (8 float4 / thread)
        #pragma unroll
        for (int i = 0; i < 8; ++i) {
            const int r = r0 + 32 * i;
            const float4 v = *(const float4*)(inp + (size_t)(blockM + r) * DMODEL + k0 + c4 * 4);
            unsigned int* p = (unsigned int*)&ash[r * LDA + c4 * 4];
            p[0] = pack_bf16x2(v.x, v.y);
            p[1] = pack_bf16x2(v.z, v.w);
        }
        // ---- B tile: 64 vocab rows x 32 K (weight is [vocab][d], contiguous in K)
        #pragma unroll
        for (int i = 0; i < 2; ++i) {
            const int r = r0 + 32 * i;
            int vr = blockN + r;
            vr = (vr < VOCAB) ? vr : (VOCAB - 1);          // clamp (no EXEC divergence)
            const float4 v = *(const float4*)(wgt + (size_t)vr * DMODEL + k0 + c4 * 4);
            unsigned int* p = (unsigned int*)&bsh[r * LDB + c4 * 4];
            p[0] = pack_bf16x2(v.x, v.y);
            p[1] = pack_bf16x2(v.z, v.w);
        }
        __syncthreads();

        // ---- fragments from LDS
        // A 16x32: lane<16 holds row=l16, K {0..7,16..23}; lane>=16 holds K {8..15,24..31}
        v16bf afrag[4];
        #pragma unroll
        for (int mi = 0; mi < 4; ++mi) {
            const int row = m_w * 64 + mi * 16 + l16;
            union { v16bf v; v8bf h[2]; } u;
            u.h[0] = *(const v8bf*)&ash[row * LDA + hp * 8];
            u.h[1] = *(const v8bf*)&ash[row * LDA + hp * 8 + 16];
            afrag[mi] = u.v;
        }
        // B 32x16: lane<16 holds col=l16, K {0..7,16..23}; lane>=16 holds K {8..15,24..31}
        v16bf bfrag[2];
        #pragma unroll
        for (int ni = 0; ni < 2; ++ni) {
            const int col = n_w * 32 + ni * 16 + l16;
            union { v16bf v; v8bf h[2]; } u;
            u.h[0] = *(const v8bf*)&bsh[col * LDB + hp * 8];
            u.h[1] = *(const v8bf*)&bsh[col * LDB + hp * 8 + 16];
            bfrag[ni] = u.v;
        }

        #pragma unroll
        for (int mi = 0; mi < 4; ++mi)
            #pragma unroll
            for (int ni = 0; ni < 2; ++ni)
                acc[mi][ni] = __builtin_amdgcn_wmma_f32_16x16x32_bf16(
                    false, afrag[mi], false, bfrag[ni],
                    (short)0, acc[mi][ni], false, false);

        __syncthreads();
    }

    // ---- store: C/D 16x16: VGPR r -> M = hp*8 + r, N = l16 (per lane)
    #pragma unroll
    for (int mi = 0; mi < 4; ++mi) {
        #pragma unroll
        for (int ni = 0; ni < 2; ++ni) {
            const int gcol = blockN + n_w * 32 + ni * 16 + l16;
            if (gcol < VOCAB) {
                const int grow = blockM + m_w * 64 + mi * 16 + hp * 8;
                float* dst = out + (size_t)grow * VOCAB + gcol;
                #pragma unroll
                for (int r = 0; r < 8; ++r)
                    dst[(size_t)r * VOCAB] = acc[mi][ni][r];
            }
        }
    }
}

// ---------------------------------------------------------------------------
// Kernel 2: per-token online logsumexp over 50257 logits (1 block / token)
// ---------------------------------------------------------------------------
__device__ __forceinline__ void lse_merge(float& m, float& s, float m2, float s2) {
    const float mm = fmaxf(m, m2);
    s = s * __expf(m - mm) + s2 * __expf(m2 - mm);
    m = mm;
}

__global__ __launch_bounds__(256)
void lse_kernel(const float* __restrict__ logits, float* __restrict__ lse)
{
    __shared__ float sm[8], ss[8];
    const int t = blockIdx.x;
    const float* row = logits + (size_t)t * VOCAB;

    float m = -3.0e38f, s = 0.0f;
    for (int i = threadIdx.x; i < VOCAB; i += 256) {
        const float x  = row[i];
        const float mm = fmaxf(m, x);
        s = s * __expf(m - mm) + __expf(x - mm);
        m = mm;
    }
    // wave32 tree reduce
    #pragma unroll
    for (int off = 16; off > 0; off >>= 1) {
        const float m2 = __shfl_xor(m, off, 32);
        const float s2 = __shfl_xor(s, off, 32);
        lse_merge(m, s, m2, s2);
    }
    if ((threadIdx.x & 31) == 0) { sm[threadIdx.x >> 5] = m; ss[threadIdx.x >> 5] = s; }
    __syncthreads();
    if (threadIdx.x == 0) {
        float M = sm[0], S = ss[0];
        #pragma unroll
        for (int w = 1; w < 8; ++w) lse_merge(M, S, sm[w], ss[w]);
        lse[t] = M + __logf(S);
    }
}

// ---------------------------------------------------------------------------
// Kernel 3: out[t, v] = logit - lse[t]   (streaming normalize)
// ---------------------------------------------------------------------------
__global__ __launch_bounds__(256)
void sub_lse_kernel(float* __restrict__ out, const float* __restrict__ lse)
{
    const int t = blockIdx.y;
    const float l = lse[t];
    float* row = out + (size_t)t * VOCAB;
    const int stride = 256 * gridDim.x;
    for (int i = blockIdx.x * 256 + threadIdx.x; i < VOCAB; i += stride)
        row[i] -= l;
}

extern "C" void kernel_launch(void* const* d_in, const int* in_sizes, int n_in,
                              void* d_out, int out_size, void* d_ws, size_t ws_size,
                              hipStream_t stream)
{
    const float* inp = (const float*)d_in[0];   // input [4096, 2048] fp32
    // d_in[1] = target (unused by the forward reference output)
    const float* wgt = (const float*)d_in[2];   // proj_weight [50257, 2048] fp32
    float* out = (float*)d_out;                 // [4096, 50257] fp32
    float* lse = (float*)d_ws;                  // 4096 floats of scratch

    dim3 g1((VOCAB + TN - 1) / TN, TOKENS / TM);   // 786 x 16
    wmma_logits_kernel<<<g1, 256, 0, stream>>>(inp, wgt, out);

    lse_kernel<<<TOKENS, 256, 0, stream>>>(out, lse);

    dim3 g3(64, TOKENS);
    sub_lse_kernel<<<g3, 256, 0, stream>>>(out, lse);
}
